// LightAttentiveConvNet_23948737642656
// MI455X (gfx1250) — matrix operand order repaired and checked by reading, not compiled, not run on hardware
//
#include <hip/hip_runtime.h>
#include <hip/hip_bf16.h>
#include <math.h>

#define B_    32
#define S_    512
#define E_    512
#define FEAT_ 256
#define ATT_  10
#define NCLS_ 8
#define KTOT_ 1536   // 3*E
#define KSTEP_ 48    // 1536/32
#define TTILES_ 32   // ceil(510/16)

typedef __attribute__((ext_vector_type(16))) __bf16 v16bf;
typedef __attribute__((ext_vector_type(8)))  __bf16 v8bf;
typedef __attribute__((ext_vector_type(8)))  float  v8f;

__device__ __forceinline__ __bf16 f2bf(float f) {
  unsigned u = __builtin_bit_cast(unsigned, f);
  u += 0x7FFFu + ((u >> 16) & 1u);               // round-to-nearest-even
  unsigned short s = (unsigned short)(u >> 16);
  return __builtin_bit_cast(__bf16, s);
}

// ---------------------------------------------------------------------------
// Kernel 1: embedding gather (+bf16 copy for WMMA) + scores (K=512,N=10) +
//           softmax over ATT. One wave32 per (b,s) row.
// ---------------------------------------------------------------------------
__global__ void k_gather_scores(const int* __restrict__ x,
                                const float* __restrict__ emb,
                                const float* __restrict__ A,
                                __bf16* __restrict__ exb,
                                float* __restrict__ attn,
                                float* __restrict__ scw) {
  int wave = blockIdx.x * 8 + (threadIdx.x >> 5);   // (b*S + s)
  int lane = threadIdx.x & 31;
  int row = x[wave];
  const float* er = emb + (size_t)row * E_;
  __bf16* eo = exb + (size_t)wave * E_;
  float p[ATT_];
#pragma unroll
  for (int a = 0; a < ATT_; ++a) p[a] = 0.f;
#pragma unroll 4
  for (int j = 0; j < 16; ++j) {
    int e = lane + 32 * j;                           // coalesced
    float v = er[e];
    eo[e] = f2bf(v);
    const float* Ar = A + e * ATT_;
#pragma unroll
    for (int a = 0; a < ATT_; ++a) p[a] += v * Ar[a];
  }
#pragma unroll
  for (int off = 16; off > 0; off >>= 1)
#pragma unroll
    for (int a = 0; a < ATT_; ++a) p[a] += __shfl_down(p[a], off);
  if (lane == 0) {
    float m = p[0];
#pragma unroll
    for (int a = 1; a < ATT_; ++a) m = fmaxf(m, p[a]);
    float ev[ATT_], sum = 0.f;
#pragma unroll
    for (int a = 0; a < ATT_; ++a) { ev[a] = __expf(p[a] - m); sum += ev[a]; }
    float inv = 1.f / sum;
    float* ao = attn + (size_t)wave * ATT_;
    float* so = scw  + (size_t)wave * ATT_;
#pragma unroll
    for (int a = 0; a < ATT_; ++a) { ao[a] = ev[a] * inv; so[a] = p[a]; }
  }
}

// ---------------------------------------------------------------------------
// Kernel 2: pre-pack Wh [FEAT,E,3] -> bf16 B-fragments in WMMA lane order.
// GEMM K index r = k*E + e. Tile (kk,fT): lane(h=lane/16,n=lane&15), element
// idx: localK = h*8 + (idx&7) + (idx>>3)*16, row r = kk*32+localK, col f = fT*16+n.
// ---------------------------------------------------------------------------
__global__ void k_pack_w(const float* __restrict__ Wh,
                         __bf16* __restrict__ Wpack) {
  int tid = blockIdx.x * blockDim.x + threadIdx.x;   // < 48*16*32*16
  int idx  = tid & 15;
  int lane = (tid >> 4) & 31;
  int tile = tid >> 9;                               // kk*16 + fT
  int kk = tile >> 4, fT = tile & 15;
  int h = lane >> 4, n = lane & 15;
  int r = kk * 32 + h * 8 + (idx & 7) + ((idx >> 3) << 4); // 0..1535
  int e = r & (E_ - 1);
  int k = r >> 9;
  int f = fT * 16 + n;
  Wpack[tid] = f2bf(Wh[((size_t)f * E_ + e) * 3 + k]);
}

// ---------------------------------------------------------------------------
// Kernel 3: P[b,f,a] = sum_t scores[b,t,a] * Wc[f,t]  (rank-10 right factor
// of the algebraically-collapsed c-branch). One wave per (b,f).
// ---------------------------------------------------------------------------
__global__ void k_P(const float* __restrict__ scw,
                    const float* __restrict__ Wc,
                    float* __restrict__ P) {
  int wave = blockIdx.x * 8 + (threadIdx.x >> 5);    // b*FEAT + f
  int lane = threadIdx.x & 31;
  int b = wave >> 8;
  int f = wave & 255;
  const float* sc = scw + (size_t)b * S_ * ATT_;
  const float* wr = Wc + (size_t)f * E_;
  float acc[ATT_];
#pragma unroll
  for (int a = 0; a < ATT_; ++a) acc[a] = 0.f;
  for (int t = lane; t < S_; t += 32) {
    float w = wr[t];
    const float* sr = sc + t * ATT_;
#pragma unroll
    for (int a = 0; a < ATT_; ++a) acc[a] += w * sr[a];
  }
#pragma unroll
  for (int off = 16; off > 0; off >>= 1)
#pragma unroll
    for (int a = 0; a < ATT_; ++a) acc[a] += __shfl_down(acc[a], off);
  if (lane == 0) {
    float* po = P + (size_t)wave * ATT_;
#pragma unroll
    for (int a = 0; a < ATT_; ++a) po[a] = acc[a];
  }
}

// ---------------------------------------------------------------------------
// Kernel 4: cmax[b,f] = max_l sum_a attn[b,l,a]*P[b,f,a]. One wave per (b,f).
// ---------------------------------------------------------------------------
__global__ void k_cmax(const float* __restrict__ attn,
                       const float* __restrict__ P,
                       float* __restrict__ cmax) {
  int wave = blockIdx.x * 8 + (threadIdx.x >> 5);
  int lane = threadIdx.x & 31;
  const float* pr = P + (size_t)wave * ATT_;
  float p[ATT_];
#pragma unroll
  for (int a = 0; a < ATT_; ++a) p[a] = pr[a];
  const float* at = attn + (size_t)(wave >> 8) * S_ * ATT_;
  float m = -INFINITY;
  for (int l = lane; l < S_; l += 32) {
    const float* ar = at + l * ATT_;
    float s = 0.f;
#pragma unroll
    for (int a = 0; a < ATT_; ++a) s += ar[a] * p[a];
    m = fmaxf(m, s);
  }
#pragma unroll
  for (int off = 16; off > 0; off >>= 1) m = fmaxf(m, __shfl_xor(m, off));
  if (lane == 0) cmax[wave] = m;
}

// ---------------------------------------------------------------------------
// Kernel 5: conv1d(k=3,valid) as implicit-im2col GEMM [510,1536]x[1536,256]
// per batch, bf16 WMMA 16x16x32 with f32 accumulate, fused max-over-t.
// One wave owns a 16(M) x 64(N) strip: 4 accumulators sharing one A fragment
// per K-step (4x A-reuse). Out-of-range M rows are load-clamped (in-bounds)
// and masked out of the max, so the K-loop is branch-free.
// ---------------------------------------------------------------------------
__global__ void k_conv_wmma(const __bf16* __restrict__ exb,
                            const __bf16* __restrict__ Wpack,
                            float* __restrict__ hpart) {
  int wave = blockIdx.x * 8 + (threadIdx.x >> 5);   // b*128 + tT*4 + fG
  int lane = threadIdx.x & 31;
  int fG = wave & 3;                                // N-group: fT = fG*4+q
  int tT = (wave >> 2) & 31;
  int b  = wave >> 7;
  int M = lane & 15;
  int h = lane >> 4;
  int t  = tT * 16 + M;                             // output time (A row)
  int tl = t < (S_ - 2) ? t : (S_ - 3);             // clamp: keep loads in-bounds
  const __bf16* arow = exb + ((size_t)b * S_ + tl) * E_ + h * 8;
  const __bf16* bbase = Wpack + ((size_t)fG * 4 * 32 + lane) * 16;
  v8f acc0 = {}, acc1 = {}, acc2 = {}, acc3 = {};
#pragma unroll 4
  for (int kk = 0; kk < KSTEP_; ++kk) {
    int Kb = kk * 32;
    int k  = Kb >> 9;                               // conv tap (const per step)
    int eb = Kb & (E_ - 1);
    // 16-bit A 16x32 layout: half h holds K = h*8+{0..7} and 16+h*8+{0..7}
    const __bf16* src = arow + (size_t)k * E_ + eb;
    v8bf c0 = *(const v8bf*)(src);
    v8bf c1 = *(const v8bf*)(src + 16);
    v16bf af = __builtin_shufflevector(c0, c1,
        0, 1, 2, 3, 4, 5, 6, 7, 8, 9, 10, 11, 12, 13, 14, 15);
    const __bf16* bp = bbase + (size_t)kk * 16 * 32 * 16;
    __builtin_prefetch(bp + 16 * 32 * 16 * 16, 0, 1);  // B frags 16 K-steps ahead
    v16bf bf0 = *(const v16bf*)(bp);
    v16bf bf1 = *(const v16bf*)(bp + 32 * 16);
    v16bf bf2 = *(const v16bf*)(bp + 2 * 32 * 16);
    v16bf bf3 = *(const v16bf*)(bp + 3 * 32 * 16);
    acc0 = __builtin_amdgcn_wmma_f32_16x16x32_bf16(false, af, false, bf0,
                                                   (short)0, acc0, false, false);
    acc1 = __builtin_amdgcn_wmma_f32_16x16x32_bf16(false, af, false, bf1,
                                                   (short)0, acc1, false, false);
    acc2 = __builtin_amdgcn_wmma_f32_16x16x32_bf16(false, af, false, bf2,
                                                   (short)0, acc2, false, false);
    acc3 = __builtin_amdgcn_wmma_f32_16x16x32_bf16(false, af, false, bf3,
                                                   (short)0, acc3, false, false);
  }
  // C/D layout: VGPR r -> M = r (lanes 0-15) or 8+r (lanes 16-31), N = lane&15
  int rb = (lane >= 16) ? 8 : 0;
  float m0 = -INFINITY, m1 = -INFINITY, m2 = -INFINITY, m3 = -INFINITY;
#pragma unroll
  for (int r = 0; r < 8; ++r) {
    int tr = tT * 16 + rb + r;
    if (tr < S_ - 2) {
      m0 = fmaxf(m0, acc0[r]);
      m1 = fmaxf(m1, acc1[r]);
      m2 = fmaxf(m2, acc2[r]);
      m3 = fmaxf(m3, acc3[r]);
    }
  }
  m0 = fmaxf(m0, __shfl_xor(m0, 16));   // merge M-halves (same N)
  m1 = fmaxf(m1, __shfl_xor(m1, 16));
  m2 = fmaxf(m2, __shfl_xor(m2, 16));
  m3 = fmaxf(m3, __shfl_xor(m3, 16));
  if (lane < 16) {
    float* hp = hpart + ((size_t)(b * TTILES_ + tT)) * FEAT_ + fG * 64 + lane;
    hp[0]  = m0;
    hp[16] = m1;
    hp[32] = m2;
    hp[48] = m3;
  }
}

// ---------------------------------------------------------------------------
// Kernel 6: reduce tile partials, tanh(h+c+bias), tiny FC [256 -> 8] + bias.
// One block per batch.
// ---------------------------------------------------------------------------
__global__ void k_final(const float* __restrict__ hpart,
                        const float* __restrict__ cmax,
                        const float* __restrict__ tb,
                        const float* __restrict__ Wfc,
                        const float* __restrict__ bfc,
                        float* __restrict__ out) {
  int b = blockIdx.x;
  int f = threadIdx.x;
  float hm = -INFINITY;
#pragma unroll 8
  for (int tT = 0; tT < TTILES_; ++tT)
    hm = fmaxf(hm, hpart[((size_t)(b * TTILES_ + tT)) * FEAT_ + f]);
  float val = tanhf(hm + cmax[b * FEAT_ + f] + tb[f]);
  __shared__ float sv[FEAT_];
  sv[f] = val;
  __syncthreads();
  if (f < NCLS_) {
    float s = bfc[f];
    const float* wr = Wfc + f * FEAT_;
#pragma unroll 8
    for (int i = 0; i < FEAT_; ++i) s += sv[i] * wr[i];
    out[b * NCLS_ + f] = s;
  }
}

// ---------------------------------------------------------------------------
extern "C" void kernel_launch(void* const* d_in, const int* in_sizes, int n_in,
                              void* d_out, int out_size, void* d_ws, size_t ws_size,
                              hipStream_t stream) {
  (void)in_sizes; (void)n_in; (void)out_size; (void)ws_size;
  const int*   x   = (const int*)d_in[0];
  const float* emb = (const float*)d_in[1];
  const float* A   = (const float*)d_in[2];
  const float* Wh  = (const float*)d_in[3];
  const float* Wc  = (const float*)d_in[4];
  const float* tb  = (const float*)d_in[5];
  const float* Wfc = (const float*)d_in[6];
  const float* bfc = (const float*)d_in[7];
  float* out = (float*)d_out;

  char* w = (char*)d_ws;
  __bf16* exb   = (__bf16*)w; w += (size_t)B_ * S_ * E_ * 2;          // 16 MB
  float*  attn  = (float*)w;  w += (size_t)B_ * S_ * ATT_ * 4;        // 640 KB
  float*  scw   = (float*)w;  w += (size_t)B_ * S_ * ATT_ * 4;        // 640 KB
  float*  P     = (float*)w;  w += (size_t)B_ * FEAT_ * ATT_ * 4;     // 320 KB
  float*  cmaxb = (float*)w;  w += (size_t)B_ * FEAT_ * 4;            // 32 KB
  __bf16* Wpack = (__bf16*)w; w += (size_t)KSTEP_ * 16 * 32 * 16 * 2; // 768 KB
  float*  hpart = (float*)w;  w += (size_t)B_ * TTILES_ * FEAT_ * 4;  // 1 MB

  k_gather_scores<<<B_ * S_ / 8, 256, 0, stream>>>(x, emb, A, exb, attn, scw);
  k_pack_w<<<(KSTEP_ * 16 * 32 * 16) / 256, 256, 0, stream>>>(Wh, Wpack);
  k_P<<<B_ * FEAT_ / 8, 256, 0, stream>>>(scw, Wc, P);
  k_cmax<<<B_ * FEAT_ / 8, 256, 0, stream>>>(attn, P, cmaxb);
  k_conv_wmma<<<B_ * TTILES_ * 4 / 8, 256, 0, stream>>>(exb, Wpack, hpart);
  k_final<<<B_, FEAT_, 0, stream>>>(hpart, cmaxb, tb, Wfc, bfc, out);
}
